// TransformerDecoder_39058432590118
// MI455X (gfx1250) — matrix operand order repaired and checked by reading, not compiled, not run on hardware
//
#include <hip/hip_runtime.h>

// ---------------------------------------------------------------------------
// Vector types
// ---------------------------------------------------------------------------
typedef __attribute__((ext_vector_type(4)))  float    f32x4;
typedef __attribute__((ext_vector_type(4)))  unsigned u32x4;
typedef __attribute__((ext_vector_type(8)))  unsigned u32x8;
typedef __attribute__((ext_vector_type(4)))  int      i32x4;
typedef __attribute__((ext_vector_type(8)))  int      i32x8;
typedef __attribute__((ext_vector_type(16))) __bf16   v16bf;
typedef __attribute__((ext_vector_type(8)))  float    v8f;

// TDM availability (probe-verified builtin; arity differs by toolchain)
#if defined(__has_builtin)
#if __has_builtin(__builtin_amdgcn_tensor_load_to_lds) && \
    __has_builtin(__builtin_amdgcn_s_wait_tensorcnt)
#define HAVE_TDM 1
#endif
#endif
#ifndef HAVE_TDM
#define HAVE_TDM 0
#endif

// Pack two fp32 into one dword of two bf16 (truncation; 2 ALU ops)
__device__ __forceinline__ unsigned pack_bf16(float lo, float hi) {
    return (__builtin_bit_cast(unsigned, hi) & 0xFFFF0000u) |
           (__builtin_bit_cast(unsigned, lo) >> 16);
}
__device__ __forceinline__ __bf16 f2bf(float f) {
    unsigned short s = (unsigned short)(__builtin_bit_cast(unsigned, f) >> 16);
    return __builtin_bit_cast(__bf16, s);
}
__device__ __forceinline__ float lrelu(float x) { return x > 0.f ? x : 0.1f * x; }

__device__ __forceinline__ float waveSum(float v) {
    for (int off = 16; off; off >>= 1) v += __shfl_xor(v, off, 32);
    return v;
}
__device__ __forceinline__ float waveMax(float v) {
    for (int off = 16; off; off >>= 1) v = fmaxf(v, __shfl_xor(v, off, 32));
    return v;
}

#if HAVE_TDM
// Issue a 2D TDM load: tile of 64 rows x 32 fp32 from row-major [.,K] matrix
// at gaddr into LDS byte offset ldsOff (dense 64x32 fp32). ISA D# layout
// (CDNA5 ISA): group0 = {count=1 | lds_addr | global_addr | type=2},
// group1 = {data_size=4B, tensor_dim0=32, tensor_dim1=64, tile_dim0=32,
//           tile_dim1=64, tensor_dim0_stride=K}.
__device__ __forceinline__ void tdm_load_tile(unsigned long long gaddr,
                                              unsigned ldsOff, int K)
{
    u32x4 g0;
    g0[0] = 1u;                                        // count=1, no gather
    g0[1] = ldsOff;                                    // lds_addr (bytes)
    g0[2] = (unsigned)gaddr;                           // global_addr[31:0]
    g0[3] = (unsigned)((gaddr >> 32) & 0x01FFFFFFull)  // global_addr[56:32]
            | 0x80000000u;                             // type=2 ("image")
    i32x8 g1;
    g1[0] = (int)(2u << 16);     // data_size = 2 -> 4 bytes
    g1[1] = (int)(32u << 16);    // tensor_dim0 = 32 (bits 63:48)
    g1[2] = (int)(64u << 16);    // tensor_dim1 = 64 (bits 95:80)
    g1[3] = (int)(32u << 16);    // tile_dim0 = 32 (bits 127:112)
    g1[4] = 64;                  // tile_dim1 = 64 (bits 143:128)
    g1[5] = K;                   // tensor_dim0_stride[31:0] (elements)
    g1[6] = 0;
    g1[7] = 0;
    i32x4 z4 = {0, 0, 0, 0};
#if __clang_major__ >= 23
    i32x8 z8 = {0, 0, 0, 0, 0, 0, 0, 0};
    __builtin_amdgcn_tensor_load_to_lds(g0, g1, z4, z4, z8, 0);
#else
    __builtin_amdgcn_tensor_load_to_lds(g0, g1, z4, z4, 0);
#endif
}
#endif

// ---------------------------------------------------------------------------
// FAST GEMM: out[M,Nc] = ACT(A[M,K] @ W[K,Nc] + bias) (+ RES ? residual)
// Requires M%64==0, K%32==0, Nc%64==0 (true for every large GEMM here).
// Block = 128 threads (4 waves), tile = 64(M) x 64(N). A tile (64x32 fp32)
// DMA'd into LDS by the Tensor Data Mover (wave0 issues, s_wait_tensorcnt,
// barrier publishes); each wave owns a 64x16 C strip -> 4 v_wmma per k-step
// sharing one B fragment. bf16 packing happens at fragment-read time.
// ---------------------------------------------------------------------------
template <int ACT, int RES>
__global__ __launch_bounds__(128) void gemm_wmma_fast(
    const float* __restrict__ A, const float* __restrict__ W,
    const float* __restrict__ bias, const float* __restrict__ residual,
    float* __restrict__ out, int M, int K, int Nc)
{
    __shared__ float lA[64 * 32];   // [m][k] fp32 tile (8KB), rows 128B

    const int t    = threadIdx.x;
    const int wave = t >> 5;
    const int lane = t & 31;
    const int half = lane >> 4;
    const int l16  = lane & 15;

    const int gm0  = blockIdx.y * 64;
    const int nCol = blockIdx.x * 64 + wave * 16 + l16;

#if !HAVE_TDM
    const int sm  = t >> 3;          // manual staging fallback indices
    const int sk4 = (t & 7) * 4;
#endif

    v8f acc[4];
#pragma unroll
    for (int s = 0; s < 4; ++s) acc[s] = (v8f){0.f,0.f,0.f,0.f,0.f,0.f,0.f,0.f};

    for (int kk = 0; kk < K; kk += 32) {
        // ---- stage A 64x32 fp32 -> LDS ----
#if HAVE_TDM
        if (wave == 0) {
            unsigned long long ga =
                (unsigned long long)(const void*)(A + (size_t)gm0 * K + kk);
            tdm_load_tile(ga, (unsigned)(size_t)(void*)lA, K);
            __builtin_amdgcn_s_wait_tensorcnt(0);
        }
#else
#pragma unroll
        for (int i = 0; i < 4; ++i) {
            int m = sm + 16 * i;
            *(f32x4*)&lA[m * 32 + sk4] =
                *(const f32x4*)(A + (size_t)(gm0 + m) * K + (kk + sk4));
        }
#endif
        __syncthreads();

        // ---- B fragment: column nCol, K run = kk + 16*half .. +15 ----
        u32x8 pb;
        {
            const float* wp = W + (size_t)(kk + 16 * half) * Nc + nCol;
            const size_t st2 = (size_t)Nc * 2;
#pragma unroll
            for (int e = 0; e < 8; ++e) {
                float lo = wp[0];
                float hi = wp[Nc];
                pb[e] = pack_bf16(lo, hi);
                wp += st2;
            }
        }
        const v16bf bfrag = __builtin_bit_cast(v16bf, pb);

        // ---- 4 M-subtiles: fp32 fragment from LDS, pack to bf16, WMMA ----
#pragma unroll
        for (int s = 0; s < 4; ++s) {
            const float* ap = &lA[(s * 16 + l16) * 32];
            f32x4 r0 = *(const f32x4*)(ap + 16 * half);       // K = 16h..16h+3
            f32x4 r1 = *(const f32x4*)(ap + 16 * half + 4);   // K = 16h+4..+7
            f32x4 r2 = *(const f32x4*)(ap + 16 * half + 8);   // K = 16h+8..
            f32x4 r3 = *(const f32x4*)(ap + 16 * half + 12);
            u32x8 pa;
            pa[0] = pack_bf16(r0[0], r0[1]); pa[1] = pack_bf16(r0[2], r0[3]);
            pa[2] = pack_bf16(r1[0], r1[1]); pa[3] = pack_bf16(r1[2], r1[3]);
            pa[4] = pack_bf16(r2[0], r2[1]); pa[5] = pack_bf16(r2[2], r2[3]);
            pa[6] = pack_bf16(r3[0], r3[1]); pa[7] = pack_bf16(r3[2], r3[3]);
            acc[s] = __builtin_amdgcn_wmma_f32_16x16x32_bf16(
                         false, __builtin_bit_cast(v16bf, pa),
                         false, bfrag, (short)0, acc[s], false, false);
        }
        __syncthreads();   // protect lA before next stage
    }

    const float bv = bias[nCol];
#pragma unroll
    for (int s = 0; s < 4; ++s) {
#pragma unroll
        for (int r = 0; r < 8; ++r) {
            int m = gm0 + s * 16 + r + 8 * half;
            float v = acc[s][r] + bv;
            if constexpr (ACT) v = lrelu(v);
            size_t idx = (size_t)m * Nc + nCol;
            if constexpr (RES) v += residual[idx];
            out[idx] = v;
        }
    }
}

// ---------------------------------------------------------------------------
// Generic GEMM for odd shapes: et1 (K=12), on2 (N=16), oe2 (N=12).
// One wave per 16x16 tile; 4 waves/block tile N. M%16==0.
// Bounds handled branch-free: addresses are clamped in-range (always-legal
// loads) and out-of-range elements zeroed with selects (v_cndmask), so the
// k-loop is straight-line load/pack/WMMA.
// ---------------------------------------------------------------------------
__global__ __launch_bounds__(128) void gemm_wmma_gen(
    const float* __restrict__ A, const float* __restrict__ W,
    const float* __restrict__ bias, const float* __restrict__ residual,
    float* __restrict__ out, int M, int K, int Nc, int act)
{
    const int wave = threadIdx.x >> 5;
    const int lane = threadIdx.x & 31;
    const int half = lane >> 4;
    const int l16  = lane & 15;

    const int n0 = (blockIdx.x * 4 + wave) * 16;
    const int m0 = blockIdx.y * 16;
    if (n0 >= ((Nc + 15) & ~15)) return;   // wave-uniform exit; no LDS used

    const int  nCol   = n0 + l16;
    const bool nok    = (nCol < Nc);
    const int  kmax   = K - 1;
    const int  nClamp = nok ? nCol : (Nc - 1);

    const float* Arow = A + (size_t)(m0 + l16) * K;
    const float* Wcol = W + nClamp;

    v8f acc = {0.f,0.f,0.f,0.f,0.f,0.f,0.f,0.f};

    for (int kk = 0; kk < K; kk += 32) {
        v16bf a;
        const int ka0 = kk + 16 * half;
        const int ka1 = kk + 16 * half + 8;
#pragma unroll
        for (int e = 0; e < 8; ++e) {
            int k0 = ka0 + e, k1 = ka1 + e;
            float v0 = Arow[k0 < K ? k0 : kmax];   // clamped, always in-bounds
            float v1 = Arow[k1 < K ? k1 : kmax];
            a[e]     = f2bf(k0 < K ? v0 : 0.f);    // select, no branch
            a[e + 8] = f2bf(k1 < K ? v1 : 0.f);
        }
        v16bf b;
        const int kb = kk + 16 * half;
#pragma unroll
        for (int e = 0; e < 16; ++e) {
            int k = kb + e;
            float v = Wcol[(size_t)(k < K ? k : kmax) * Nc];
            b[e] = f2bf((nok && k < K) ? v : 0.f);
        }
        acc = __builtin_amdgcn_wmma_f32_16x16x32_bf16(
                  false, a, false, b, (short)0, acc, false, false);
    }

    if (!nok) return;
    const float bv = bias[nCol];
#pragma unroll
    for (int r = 0; r < 8; ++r) {
        int m = m0 + r + 8 * half;
        float v = acc[r] + bv;
        if (act) v = lrelu(v);
        size_t idx = (size_t)m * Nc + nCol;
        if (residual) v += residual[idx];
        out[idx] = v;
    }
}

// ---------------------------------------------------------------------------
// LayerNorm over rows of 256 (one block per row, 8 waves), optional lrelu.
// ---------------------------------------------------------------------------
__global__ __launch_bounds__(256) void layernorm_kernel(
    const float* __restrict__ in, const float* __restrict__ gamma,
    const float* __restrict__ beta, float* __restrict__ out, int act)
{
    const int row  = blockIdx.x;
    const int t    = threadIdx.x;
    const int lane = t & 31, wave = t >> 5;
    __shared__ float red[8];

    float x = in[(size_t)row * 256 + t];
    float s = waveSum(x);
    if (lane == 0) red[wave] = s;
    __syncthreads();
    float mean = 0.f;
#pragma unroll
    for (int i = 0; i < 8; ++i) mean += red[i];
    mean *= (1.f / 256.f);
    __syncthreads();

    float d = x - mean;
    float v = waveSum(d * d);
    if (lane == 0) red[wave] = v;
    __syncthreads();
    float var = 0.f;
#pragma unroll
    for (int i = 0; i < 8; ++i) var += red[i];
    var *= (1.f / 256.f);

    float y = gamma[t] * d * rsqrtf(var + 1e-5f) + beta[t];
    if (act) y = lrelu(y);
    out[(size_t)row * 256 + t] = y;
}

// ---------------------------------------------------------------------------
// Edge-aware MHA core. Block = (b,i); wave h == head h, lane == d.
// Writes NE = lrelu(new_edges) and WV = softmax-weighted V. NE may alias EA.
// ---------------------------------------------------------------------------
__global__ __launch_bounds__(256) void attention_kernel(
    const float* __restrict__ Q, const float* __restrict__ Kx,
    const float* __restrict__ V, const float* __restrict__ EM,
    const float* __restrict__ EA, float* __restrict__ NE,
    float* __restrict__ WV)
{
    const int bi   = blockIdx.x;        // b*32 + i
    const int t    = threadIdx.x;       // channel = h*32 + d
    const int h    = t >> 5;
    const int lane = t & 31;            // = d
    const size_t nb = (size_t)(bi & ~31);
    const size_t ebase = (size_t)bi * 32 * 256;

    __shared__ float sh[32 * 8];        // [j*8 + h]

    const float qv = Q[(size_t)bi * 256 + t] * (1.f / 16.f);  // /sqrt(256)

    for (int j = 0; j < 32; ++j) {
        float attn = qv * Kx[(nb + j) * 256 + t];
        size_t eidx = ebase + (size_t)j * 256 + t;
        float ne = attn * EM[eidx] + attn + EA[eidx];
        float s  = waveSum(ne);
        if (lane == 0) sh[j * 8 + h] = s;
        NE[eidx] = lrelu(ne);
    }
    __syncthreads();

    {   // softmax over j: wave h handles head h, lane index = j
        float x  = sh[lane * 8 + h];
        float mx = waveMax(x);
        float ex = expf(x - mx);
        float sm = waveSum(ex);
        sh[lane * 8 + h] = ex / sm;
    }
    __syncthreads();

    float acc = 0.f;
    for (int j = 0; j < 32; ++j)
        acc += sh[j * 8 + h] * V[(nb + j) * 256 + t];
    WV[(size_t)bi * 256 + t] = acc;
}

// ---------------------------------------------------------------------------
// Output heads
// ---------------------------------------------------------------------------
__global__ void node_head_kernel(const float* __restrict__ in,
                                 float* __restrict__ out, int rows)
{
    int row = blockIdx.x * blockDim.x + threadIdx.x;
    if (row >= rows) return;
    const float* r = in + (size_t)row * 16;
    float* o = out + (size_t)row * 16;
    o[0] = 1.f / (1.f + expf(-r[0]));
    float mx = r[1];
#pragma unroll
    for (int c = 2; c < 6; ++c) mx = fmaxf(mx, r[c]);
    float e[5], s = 0.f;
#pragma unroll
    for (int c = 0; c < 5; ++c) { e[c] = expf(r[1 + c] - mx); s += e[c]; }
#pragma unroll
    for (int c = 0; c < 5; ++c) o[1 + c] = e[c] / s;
#pragma unroll
    for (int c = 6; c < 16; ++c) o[c] = r[c];
}

__global__ void edge_head_kernel(const float* __restrict__ in,
                                 float* __restrict__ out, int rows)
{
    int row = blockIdx.x * blockDim.x + threadIdx.x;
    if (row >= rows) return;
    const float* r = in + (size_t)row * 12;
    float* o = out + (size_t)row * 12;
#pragma unroll
    for (int g = 0; g < 3; ++g) {
        const float* rr = r + g * 4;
        float mx = fmaxf(fmaxf(rr[0], rr[1]), fmaxf(rr[2], rr[3]));
        float e0 = expf(rr[0] - mx), e1 = expf(rr[1] - mx);
        float e2 = expf(rr[2] - mx), e3 = expf(rr[3] - mx);
        float s = e0 + e1 + e2 + e3;
        o[g * 4 + 0] = e0 / s; o[g * 4 + 1] = e1 / s;
        o[g * 4 + 2] = e2 / s; o[g * 4 + 3] = e3 / s;
    }
}

// ---------------------------------------------------------------------------
// Host-side orchestration
// ---------------------------------------------------------------------------
static void gemm(hipStream_t s, const float* A, const float* W, const float* bias,
                 const float* res, float* out, int M, int K, int Nc, int act)
{
    if ((M % 64 == 0) && (K % 32 == 0) && (Nc % 64 == 0)) {
        dim3 g(Nc / 64, M / 64), b(128);
        if (res)
            gemm_wmma_fast<0, 1><<<g, b, 0, s>>>(A, W, bias, res, out, M, K, Nc);
        else if (act)
            gemm_wmma_fast<1, 0><<<g, b, 0, s>>>(A, W, bias, res, out, M, K, Nc);
        else
            gemm_wmma_fast<0, 0><<<g, b, 0, s>>>(A, W, bias, res, out, M, K, Nc);
    } else {
        dim3 g((Nc + 63) / 64, M / 16), b(128);
        gemm_wmma_gen<<<g, b, 0, s>>>(A, W, bias, res, out, M, K, Nc, act);
    }
}

extern "C" void kernel_launch(void* const* d_in, const int* in_sizes, int n_in,
                              void* d_out, int out_size, void* d_ws, size_t ws_size,
                              hipStream_t stream)
{
    (void)in_sizes; (void)n_in; (void)out_size; (void)ws_size;
    auto P = [&](int i) { return (const float*)d_in[i]; };

    // --- workspace bump allocator (fp32 buffers, 256B aligned) ---
    char* base = (char*)d_ws;
    size_t off = 0;
    auto alloc = [&](size_t nfl) {
        float* p = (float*)(base + off);
        off = (off + nfl * sizeof(float) + 255) & ~(size_t)255;
        return p;
    };
    const size_t NODE = 2048u * 256u;          // (B*32, 256)
    const size_t EDGE = 65536u * 256u;         // (B*1024, 256)

    float* bufA = alloc(64u * 4096u);          // cn1 out
    float* bufB = alloc(64u * 4096u);          // cn2 out == nodes0 (2048,128)
    float* bufC = alloc(64u * 12288u);         // ce1 out
    float* bufD = alloc(64u * 12288u);         // ce2 out == edges0 (65536,12); reused for oe2 out
    float* N0 = alloc(NODE); float* N1 = alloc(NODE); float* N2 = alloc(NODE);
    float* N3 = alloc(NODE); float* N4 = alloc(NODE); float* N5 = alloc(NODE);
    float* N6 = alloc(NODE); float* N7 = alloc(NODE);
    float* Nsm = alloc(2048u * 16u);
    float* E0 = alloc(EDGE); float* E1 = alloc(EDGE);
    float* E2 = alloc(EDGE); float* E3 = alloc(EDGE);

    // Input leaf order (jax tree: sorted dict keys; tuples (w,b) in order):
    // 0 latents | 1,2 ce1 | 3,4 ce2 | 5,6 cn1 | 7,8 cn2 | 9,10 et1 | 11,12 et2
    // 13.. layers[l] (30 leaves: add,eo,k,ln_e,ln_e2,ln_n,ln_n2,
    //       mlp_e1,mlp_e2,mlp_n1,mlp_n2,mul,no,q,v)
    // 133,134 nt1 | 135,136 nt2 | 137,138 oe1 | 139,140 oe2 | 141,142 on1 | 143,144 on2

    // ---------------- Stage A: decode MLPs ----------------
    gemm(stream, P(0),  P(5),  P(6),  nullptr, bufA, 64, 512,   4096,  1);  // cn1+lrelu
    gemm(stream, bufA,  P(7),  P(8),  nullptr, bufB, 64, 4096,  4096,  1);  // cn2+lrelu
    gemm(stream, P(0),  P(1),  P(2),  nullptr, bufC, 64, 512,   12288, 1);  // ce1+lrelu
    gemm(stream, bufC,  P(3),  P(4),  nullptr, bufD, 64, 12288, 12288, 1);  // ce2+lrelu
    gemm(stream, bufB,  P(133),P(134),nullptr, N1,   2048, 128, 256, 1);    // nt1+lrelu
    gemm(stream, N1,    P(135),P(136),nullptr, N0,   2048, 256, 256, 0);    // nt2
    gemm(stream, bufD,  P(9),  P(10), nullptr, E1,   65536, 12,  256, 1);   // et1+lrelu (generic)
    gemm(stream, E1,    P(11), P(12), nullptr, E0,   65536, 256, 256, 0);   // et2

    // ---------------- Stage B: 4 transformer layers ----------------
    for (int l = 0; l < 4; ++l) {
        const int q = 13 + l * 30;
        layernorm_kernel<<<2048, 256, 0, stream>>>(N0, P(q+10), P(q+11), N1, 0);   // ln_n
        layernorm_kernel<<<65536,256, 0, stream>>>(E0, P(q+6),  P(q+7),  E1, 0);   // ln_e
        gemm(stream, N1, P(q+26), P(q+27), nullptr, N2, 2048, 256, 256, 0);        // q
        gemm(stream, N1, P(q+4),  P(q+5),  nullptr, N3, 2048, 256, 256, 0);        // k
        gemm(stream, N1, P(q+28), P(q+29), nullptr, N4, 2048, 256, 256, 0);        // v
        gemm(stream, E1, P(q+22), P(q+23), nullptr, E2, 65536, 256, 256, 0);       // mul
        gemm(stream, E1, P(q+0),  P(q+1),  nullptr, E3, 65536, 256, 256, 0);       // add
        attention_kernel<<<2048, 256, 0, stream>>>(N2, N3, N4, E2, E3, E3, N5);    // NE->E3, WV->N5
        gemm(stream, N5, P(q+24), P(q+25), N0, N6, 2048, 256, 256, 0);             // no + nodes -> an
        gemm(stream, E3, P(q+2),  P(q+3),  E0, E0, 65536, 256, 256, 0);            // eo + edges -> ae
        layernorm_kernel<<<2048, 256, 0, stream>>>(N6, P(q+12), P(q+13), N7, 1);   // ln_n2 + lrelu
        gemm(stream, N7, P(q+18), P(q+19), nullptr, N1, 2048, 256, 256, 1);        // mlp_n1+lrelu
        gemm(stream, N1, P(q+20), P(q+21), N6, N0, 2048, 256, 256, 0);             // mlp_n2 + an
        layernorm_kernel<<<65536,256, 0, stream>>>(E0, P(q+8), P(q+9), E1, 0);     // ln_e2
        gemm(stream, E1, P(q+14), P(q+15), nullptr, E2, 65536, 256, 256, 1);       // mlp_e1+lrelu
        gemm(stream, E2, P(q+16), P(q+17), E0, E0, 65536, 256, 256, 0);            // mlp_e2 + ae
    }

    // ---------------- Stage C: output heads ----------------
    gemm(stream, N0, P(141), P(142), nullptr, N1,   2048, 256, 256, 1);   // on1+lrelu
    gemm(stream, N1, P(143), P(144), nullptr, Nsm,  2048, 256, 16,  0);   // on2 (generic)
    gemm(stream, E0, P(137), P(138), nullptr, E1,   65536, 256, 256, 1);  // oe1+lrelu
    gemm(stream, E1, P(139), P(140), nullptr, bufD, 65536, 256, 12,  0);  // oe2 (generic)

    float* out_nodes = (float*)d_out;
    float* out_edges = out_nodes + 2048 * 16;
    node_head_kernel<<<8,   256, 0, stream>>>(Nsm,  out_nodes, 2048);
    edge_head_kernel<<<256, 256, 0, stream>>>(bufD, out_edges, 65536);
}
